// TransformerMultiHeadAttention_5892695130551
// MI455X (gfx1250) — matrix operand order repaired and checked by reading, not compiled, not run on hardware
//
#include <hip/hip_runtime.h>

typedef __attribute__((ext_vector_type(16))) _Float16 v16h;
typedef __attribute__((ext_vector_type(8)))  float    v8f;
typedef __attribute__((ext_vector_type(4)))  int      v4i;

#define D_MODEL 1024
#define N_HEADS 16
#define DEPTH   64
#define NEG_BIG -1.0e9f

#if defined(__has_builtin)
#if __has_builtin(__builtin_amdgcn_global_store_async_from_lds_b128)
#define HAVE_ASYNC_STORE 1
#endif
#endif
#ifndef HAVE_ASYNC_STORE
#define HAVE_ASYNC_STORE 0
#endif

// AS-qualified int4 pointer types for the async LDS->global store builtin
typedef __attribute__((address_space(1))) v4i g_v4i;
typedef __attribute__((address_space(3))) v4i l_v4i;

// ---------------- WMMA helpers ----------------

__device__ inline v8f wmma32(v16h a, v16h b, v8f c) {
  // D = A(16x32 f16) * B(32x16 f16) + C(16x16 f32)
  return __builtin_amdgcn_wmma_f32_16x16x32_f16(
      false, a, false, b, (short)0, c, false, false);
}

__device__ inline v8f v8f_zero() {
  v8f z;
#pragma unroll
  for (int i = 0; i < 8; ++i) z[i] = 0.0f;
  return z;
}

// Two 16-byte chunks -> one v16h fragment (forces 2x b128 loads)
__device__ inline v16h ld2x128_f16(const _Float16* p0, const _Float16* p1) {
  union { v4i i[2]; v16h h; } u;
  u.i[0] = *(const v4i*)p0;
  u.i[1] = *(const v4i*)p1;
  return u.h;
}

// A fragment (16x32 f16): lane row = lane&15; K chunks at +0/8 and +16 halves.
__device__ inline v16h frag_a_f16(const _Float16* p, int lane) {
  p += (lane >> 4) << 3;
  return ld2x128_f16(p, p + 16);
}
// B fragment (32x16 f16): lane col = lane&15; K chunks at +0/8 and +16 halves.
__device__ inline v16h frag_b_contig_f16(const _Float16* p, int lane) {
  p += (lane >> 4) << 3;
  return ld2x128_f16(p, p + 16);
}

// A fragment from f32 data (global or LDS): four float4 loads + pack-convert.
__device__ inline v16h frag_a_f32(const float* p, int lane) {
  p += (lane >> 4) << 3;
  const float4 x0 = *(const float4*)(p);
  const float4 x1 = *(const float4*)(p + 4);
  const float4 y0 = *(const float4*)(p + 16);
  const float4 y1 = *(const float4*)(p + 20);
  v16h a;
  a[0]  = (_Float16)x0.x; a[1]  = (_Float16)x0.y;
  a[2]  = (_Float16)x0.z; a[3]  = (_Float16)x0.w;
  a[4]  = (_Float16)x1.x; a[5]  = (_Float16)x1.y;
  a[6]  = (_Float16)x1.z; a[7]  = (_Float16)x1.w;
  a[8]  = (_Float16)y0.x; a[9]  = (_Float16)y0.y;
  a[10] = (_Float16)y0.z; a[11] = (_Float16)y0.w;
  a[12] = (_Float16)y1.x; a[13] = (_Float16)y1.y;
  a[14] = (_Float16)y1.z; a[15] = (_Float16)y1.w;
  return a;
}

// ---------------- Kernel 0: weight transpose + f32->f16 convert ----------------
__global__ __launch_bounds__(256)
void wt_kernel(const float* __restrict__ W, _Float16* __restrict__ Wt) {
  __shared__ _Float16 t[32][33];
  const int tx = threadIdx.x & 31, ty = threadIdx.x >> 5;   // 32 x 8
  const int bk = blockIdx.x * 32, bn = blockIdx.y * 32;
#pragma unroll
  for (int i = 0; i < 32; i += 8)
    t[ty + i][tx] = (_Float16)W[(size_t)(bk + ty + i) * D_MODEL + bn + tx];
  __syncthreads();
#pragma unroll
  for (int i = 0; i < 32; i += 8)
    Wt[(size_t)(bn + ty + i) * D_MODEL + bk + tx] = t[tx][ty + i];
}

// ---------------- Kernel 1: projection GEMM ----------------
// C = (X(f32) @ W + bias) * scale, W transposed f16 (N x K); 16x64 strip/wave.
// TRANSPOSED=0: out[((b*H+h)*S + s)*DEPTH + d]  (head-major, Q/K; scalar stores)
// TRANSPOSED=1: out[((b*H+h)*DEPTH + d)*S + s]  (depth-major, V; b128 stores)
template <int TRANSPOSED>
__global__ __launch_bounds__(32)
void proj_gemm_kernel(const float* __restrict__ X, const _Float16* __restrict__ Wt,
                      const float* __restrict__ bias, _Float16* __restrict__ out,
                      int S, float scale) {
  const int lane = threadIdx.x;
  const int mt = blockIdx.x, nt = blockIdx.y;
  const int c15 = lane & 15;
  const int row = mt * 16 + c15;
  const float* rowp = X + (size_t)row * D_MODEL;

  v8f acc[4];
#pragma unroll
  for (int t = 0; t < 4; ++t) acc[t] = v8f_zero();

  for (int kb = 0; kb < D_MODEL; kb += 32) {
    v16h a = frag_a_f32(rowp + kb, lane);
#pragma unroll
    for (int t = 0; t < 4; ++t) {
      const _Float16* bp = Wt + (size_t)(nt * 64 + t * 16 + c15) * D_MODEL + kb;
      acc[t] = wmma32(a, frag_b_contig_f16(bp, lane), acc[t]);
    }
  }

  const int m0 = mt * 16 + ((lane >> 4) << 3);   // 8-run stays in one batch (S%16==0)
  const int bi = (m0 >= S) ? 1 : 0;              // B == 2, avoid div/mod
  const int s0 = m0 - (bi ? S : 0);

#pragma unroll
  for (int t = 0; t < 4; ++t) {
    const int col = nt * 64 + t * 16 + c15;
    const float bn = bias[col];
    const int h = col >> 6, d = col & 63;
    if (TRANSPOSED) {
      union { v4i i; _Float16 hh[8]; } u;
#pragma unroll
      for (int r = 0; r < 8; ++r) u.hh[r] = (_Float16)((acc[t][r] + bn) * scale);
      *(v4i*)(out + (((size_t)(bi * N_HEADS + h)) * DEPTH + d) * S + s0) = u.i;
    } else {
#pragma unroll
      for (int r = 0; r < 8; ++r)
        out[(((size_t)(bi * N_HEADS + h)) * S + (s0 + r)) * DEPTH + d] =
            (_Float16)((acc[t][r] + bn) * scale);
    }
  }
}

// ---------------- Kernel 2: attention (logits + softmax + P@V) ----------------
// One wave per (b, h, 16-row q tile).
// LDS: 16 x S f32 logits/probs strip (128 KB) + 16 x S f16 prob mirror (64 KB).
__global__ __launch_bounds__(32)
void attn_kernel(const _Float16* __restrict__ Qh, const _Float16* __restrict__ Kh,
                 const _Float16* __restrict__ Vt, const int* __restrict__ mask,
                 float* __restrict__ attn, _Float16* __restrict__ ctx, int S) {
  extern __shared__ float lds[];                       // 16*S f32
  _Float16* ldsh = (_Float16*)(lds + 16 * S);          // 16*S f16
  const int lane = threadIdx.x;
  const int qt = blockIdx.x, h = blockIdx.y, b = blockIdx.z;
  const int c15 = lane & 15;
  const _Float16* Q = Qh + ((size_t)(b * N_HEADS + h)) * S * DEPTH;
  const _Float16* K = Kh + ((size_t)(b * N_HEADS + h)) * S * DEPTH;
  const _Float16* V = Vt + ((size_t)(b * N_HEADS + h)) * DEPTH * S;  // (depth, S)

  // Q tile A-fragments (Q pre-scaled by 1/sqrt(depth))
  const int rowq = qt * 16 + c15;
  v16h aq0 = frag_a_f16(Q + (size_t)rowq * DEPTH, lane);
  v16h aq1 = frag_a_f16(Q + (size_t)rowq * DEPTH + 32, lane);

  // ---- logits = Q K^T + mask * NEG_BIG -> LDS (f32) ----
  const int mbase = b * S;
  for (int sk = 0; sk < S; sk += 16) {
    const _Float16* kp = K + (size_t)(sk + c15) * DEPTH;   // row sk+col of Kh
    v8f acc = v8f_zero();
    acc = wmma32(aq0, frag_b_contig_f16(kp, lane), acc);
    acc = wmma32(aq1, frag_b_contig_f16(kp + 32, lane), acc);
    const int coln = sk + c15;
    const float mval = (float)mask[mbase + coln] * NEG_BIG;
#pragma unroll
    for (int r = 0; r < 8; ++r) {
      int rm = ((lane >> 4) << 3) + r;
      lds[rm * S + coln] = acc[r] + mval;
    }
  }
  __syncthreads();

  // ---- softmax per row (two lanes per row); write f32 + f16 prob copies ----
  {
    const int row = lane & 15;
    const int off = (lane >> 4) * (S / 2);
    float* rp = lds + row * S + off;
    _Float16* hp = ldsh + row * S + off;
    float mx = -3.0e38f;
    for (int c = 0; c < S / 2; ++c) mx = fmaxf(mx, rp[c]);
    mx = fmaxf(mx, __shfl_xor(mx, 16, 32));
    float sum = 0.0f;
    for (int c = 0; c < S / 2; ++c) sum += __expf(rp[c] - mx);
    sum += __shfl_xor(sum, 16, 32);
    const float inv = 1.0f / sum;
    for (int c = 0; c < S / 2; ++c) {
      float pv = __expf(rp[c] - mx) * inv;
      rp[c] = pv;
      hp[c] = (_Float16)pv;
    }
  }
  __syncthreads();

  // ---- attention-probability writeout: async LDS -> global (b128/lane) ----
  float* ap = attn + (((size_t)(b * N_HEADS + h)) * S + qt * 16) * S;
#if HAVE_ASYNC_STORE
  for (int i = lane * 4; i < 16 * S; i += 32 * 4) {
    __builtin_amdgcn_global_store_async_from_lds_b128(
        (g_v4i*)(ap + i), (l_v4i*)(lds + i), 0, 0);
  }
#if __has_builtin(__builtin_amdgcn_s_wait_asynccnt)
  __builtin_amdgcn_s_wait_asynccnt(0);
#endif
#else
  for (int i = lane; i < 16 * S; i += 32) ap[i] = lds[i];
#endif

  // ---- ctx = P @ V : contract over S, 4 N-chunks cover depth 64 ----
  v8f accv[4];
#pragma unroll
  for (int nc = 0; nc < 4; ++nc) accv[nc] = v8f_zero();

  for (int sk = 0; sk < S; sk += 32) {
    v16h a = frag_a_f16(ldsh + (size_t)c15 * S + sk, lane);   // 2x ds_load_b128
#pragma unroll
    for (int nc = 0; nc < 4; ++nc) {
      const _Float16* vp = V + (size_t)(nc * 16 + c15) * S + sk;  // contiguous in K
      accv[nc] = wmma32(a, frag_b_contig_f16(vp, lane), accv[nc]);
    }
  }

#pragma unroll
  for (int nc = 0; nc < 4; ++nc) {
#pragma unroll
    for (int r = 0; r < 8; ++r) {
      int s = qt * 16 + ((lane >> 4) << 3) + r;
      int d = nc * 16 + c15;
      ctx[((size_t)(b * S + s)) * D_MODEL + h * DEPTH + d] = (_Float16)accv[nc][r];
    }
  }
}

// ---------------- Kernel 3: output GEMM ----------------
// out = ctx(f16, M x 1024) @ Wo + bo -> f32, Wo transposed f16.
__global__ __launch_bounds__(32)
void out_gemm_kernel(const _Float16* __restrict__ A, const _Float16* __restrict__ Wt,
                     const float* __restrict__ bias, float* __restrict__ out) {
  const int lane = threadIdx.x;
  const int mt = blockIdx.x, nt = blockIdx.y;
  const int c15 = lane & 15;
  const int row = mt * 16 + c15;
  const _Float16* rowp = A + (size_t)row * D_MODEL;

  v8f acc[4];
#pragma unroll
  for (int t = 0; t < 4; ++t) acc[t] = v8f_zero();

  for (int kb = 0; kb < D_MODEL; kb += 32) {
    v16h a = frag_a_f16(rowp + kb, lane);
#pragma unroll
    for (int t = 0; t < 4; ++t) {
      const _Float16* bp = Wt + (size_t)(nt * 64 + t * 16 + c15) * D_MODEL + kb;
      acc[t] = wmma32(a, frag_b_contig_f16(bp, lane), acc[t]);
    }
  }
#pragma unroll
  for (int t = 0; t < 4; ++t) {
    const int col = nt * 64 + t * 16 + c15;
    const float bn = bias[col];
#pragma unroll
    for (int r = 0; r < 8; ++r) {
      int m = mt * 16 + ((lane >> 4) << 3) + r;
      out[(size_t)m * D_MODEL + col] = acc[t][r] + bn;
    }
  }
}

// ---------------- launch ----------------
extern "C" void kernel_launch(void* const* d_in, const int* in_sizes, int n_in,
                              void* d_out, int out_size, void* d_ws, size_t ws_size,
                              hipStream_t stream) {
  const float* v    = (const float*)d_in[0];
  const float* k    = (const float*)d_in[1];
  const float* q    = (const float*)d_in[2];
  const int*   mask = (const int*)  d_in[3];
  const float* wq   = (const float*)d_in[4];
  const float* wk   = (const float*)d_in[5];
  const float* wv   = (const float*)d_in[6];
  const float* wo   = (const float*)d_in[7];
  const float* bq   = (const float*)d_in[8];
  const float* bk   = (const float*)d_in[9];
  const float* bv   = (const float*)d_in[10];
  const float* bo   = (const float*)d_in[11];

  const int B = 2;
  const int S = in_sizes[3] / B;        // mask has B*S elements
  const int M = B * S;

  // workspace: WqT|WkT|WvT|WoT (2MB each) then Qh|Kh|Vt|ctx (8MB each) = 40MB
  const size_t wElems = (size_t)D_MODEL * D_MODEL;
  const size_t headElems = (size_t)M * D_MODEL;
  _Float16* WqT = (_Float16*)d_ws;
  _Float16* WkT = WqT + wElems;
  _Float16* WvT = WkT + wElems;
  _Float16* WoT = WvT + wElems;
  _Float16* Qh  = WoT + wElems;
  _Float16* Kh  = Qh + headElems;
  _Float16* Vt  = Kh + headElems;
  _Float16* ctx = Vt + headElems;

  float* outp = (float*)d_out;
  float* attn = outp + (size_t)M * D_MODEL;

  dim3 tg(D_MODEL / 32, D_MODEL / 32);
  wt_kernel<<<tg, 256, 0, stream>>>(wq, WqT);
  wt_kernel<<<tg, 256, 0, stream>>>(wk, WkT);
  wt_kernel<<<tg, 256, 0, stream>>>(wv, WvT);
  wt_kernel<<<tg, 256, 0, stream>>>(wo, WoT);

  dim3 pg(M / 16, D_MODEL / 64);
  proj_gemm_kernel<0><<<pg, 32, 0, stream>>>(q, WqT, bq, Qh, S, 0.125f); // fold 1/sqrt(64)
  proj_gemm_kernel<0><<<pg, 32, 0, stream>>>(k, WkT, bk, Kh, S, 1.0f);
  proj_gemm_kernel<1><<<pg, 32, 0, stream>>>(v, WvT, bv, Vt, S, 1.0f);   // depth-major V

  // 16*S f32 (128 KB) + 16*S f16 (64 KB) = 192 KB of 320 KB/WGP
  const size_t ldsBytes = (size_t)16 * S * (sizeof(float) + sizeof(_Float16));
  attn_kernel<<<dim3(S / 16, N_HEADS, B), 32, ldsBytes, stream>>>(
      Qh, Kh, Vt, mask, attn, ctx, S);

  out_gemm_kernel<<<pg, 32, 0, stream>>>(ctx, WoT, bo, outp);
}
